// CrossAttention_23038204575762
// MI455X (gfx1250) — compile-verified, matrix operand order checked by reading, not run on hardware
//
#include <hip/hip_runtime.h>
#include <hip/hip_bf16.h>

typedef __attribute__((ext_vector_type(16))) _Float16 v16h;
typedef __attribute__((ext_vector_type(8)))  _Float16 v8h;
typedef __attribute__((ext_vector_type(8)))  float    v8f;

// ---------------------------------------------------------------------------
// f32 -> f16 cast
// ---------------------------------------------------------------------------
__global__ void cast_f32_to_f16(const float* __restrict__ in,
                                _Float16* __restrict__ out, long n) {
  long i = (long)blockIdx.x * blockDim.x + threadIdx.x;
  if (i < n) out[i] = (_Float16)in[i];
}

// ---------------------------------------------------------------------------
// Strided batched GEMM: C[m,n] = sum_k A[m,k]*B[k,n] (+bias +residual +beta*C)
//
// Layout contract (all call sites arrange data to satisfy this):
//   A: K is ALWAYS unit-stride (a_cs == 1), row stride a_rs.
//   B: if BKC, K unit-stride (b_rs == 1), col stride b_cs;
//      else    N unit-stride (b_cs == 1), row stride b_rs.
// All base pointers / batch strides / row strides are 8-element (16B) aligned,
// so staging is one global_load_b128 (+ one ds_store_b128 or 8x ds_store_b16)
// per thread per 32x32 tile — single-wait, fully vectorized.
//
// Batch b = blockIdx.z splits as bo=b/inner, bi=b%inner with independent
// strides (handles (n), (n,y) and (n,x) batch geometries).
// Block: 128 thr = 4 waves; block tile 32x32; wave tile 16x16; K-step 32
// via v_wmma_f32_16x16x32_f16 (f32 accumulate).
// ---------------------------------------------------------------------------
template<bool BKC>
__global__ __launch_bounds__(128)
void wmma_gemm_f16(const _Float16* __restrict__ A, long a_rs, long a_bso, long a_bsi,
                   const _Float16* __restrict__ B, long b_rs, long b_cs, long b_bso, long b_bsi,
                   void* __restrict__ C, long c_rs, long c_cs, long c_bso, long c_bsi,
                   int K, int inner_count,
                   const float* __restrict__ bias, const float* __restrict__ residual,
                   int c_is_f32, int beta)
{
  __shared__ _Float16 As[32][32];   // [m][k], rows 64B => 16B-aligned b128
  __shared__ _Float16 Bs[32][32];   // [n][k]

  const int tid  = threadIdx.x;
  const int lane = tid & 31;
  const int wave = tid >> 5;
  const int wm   = (wave & 1) << 4;
  const int wn   = (wave >> 1) << 4;
  const int m0   = blockIdx.y * 32;
  const int n0   = blockIdx.x * 32;
  const int bo   = blockIdx.z / inner_count;
  const int bi   = blockIdx.z % inner_count;

  const _Float16* Ab = A + (long)bo * a_bso + (long)bi * a_bsi;
  const _Float16* Bb = B + (long)bo * b_bso + (long)bi * b_bsi;
  const long coff = (long)bo * c_bso + (long)bi * c_bsi;

  const int hsel = lane >> 4;     // K-half select per ISA 16-bit A/B layout
  const int mr   = lane & 15;
  const int sm   = tid >> 2;      // staging row   (0..31)
  const int sk8  = (tid & 3) << 3;// staging 8-col (0,8,16,24)

  v8f acc = {};

  for (int k0 = 0; k0 < K; k0 += 32) {
    __syncthreads();
    // --- vectorized staging: both 16B loads issued before any LDS store ---
    v8h va = *(const v8h*)(Ab + (long)(m0 + sm) * a_rs + (k0 + sk8));
    v8h vb;
    if (BKC) vb = *(const v8h*)(Bb + (long)(n0 + sm) * b_cs + (k0 + sk8));
    else     vb = *(const v8h*)(Bb + (long)(k0 + sm) * b_rs + (n0 + sk8));
    *(v8h*)&As[sm][sk8] = va;
    if (BKC) {
      *(v8h*)&Bs[sm][sk8] = vb;
    } else {
#pragma unroll
      for (int j = 0; j < 8; ++j) Bs[sk8 + j][sm] = vb[j];
    }
    __syncthreads();

    if (k0 + 32 < K) {   // next K tile toward L2 (global_prefetch_b8)
      __builtin_prefetch(Ab + (long)(m0 + sm) * a_rs + (k0 + 32 + sk8), 0, 1);
      if (BKC) __builtin_prefetch(Bb + (long)(n0 + sm) * b_cs + (k0 + 32 + sk8), 0, 1);
      else     __builtin_prefetch(Bb + (long)(k0 + 32 + sm) * b_rs + (n0 + sk8), 0, 1);
    }

    // 16-bit A (16x32): lane<16 -> M=lane, K in {0..7,16..23};
    // lane>=16 -> M=lane-16, K in {8..15,24..31}. B (32x16) mirrors with N.
    v8h alo = *(const v8h*)&As[wm + mr][hsel * 8];
    v8h ahi = *(const v8h*)&As[wm + mr][16 + hsel * 8];
    v8h blo = *(const v8h*)&Bs[wn + mr][hsel * 8];
    v8h bhi = *(const v8h*)&Bs[wn + mr][16 + hsel * 8];
    v16h av = __builtin_shufflevector(alo, ahi, 0,1,2,3,4,5,6,7,8,9,10,11,12,13,14,15);
    v16h bv = __builtin_shufflevector(blo, bhi, 0,1,2,3,4,5,6,7,8,9,10,11,12,13,14,15);
    acc = __builtin_amdgcn_wmma_f32_16x16x32_f16(false, av, false, bv,
                                                 (short)0, acc, false, false);
  }

  // C/D layout: VGPR r, lane l -> M = r + 8*(l>>4), N = l&15.
#pragma unroll
  for (int r = 0; r < 8; ++r) {
    int ml = m0 + wm + r + (hsel << 3);
    int nl = n0 + wn + mr;
    long off = coff + (long)ml * c_rs + (long)nl * c_cs;
    float v = acc[r];
    if (bias)     v += bias[ml];
    if (residual) v += residual[off];
    if (c_is_f32) {
      float* p = (float*)C + off;
      if (beta) v += *p;
      *p = v;
    } else {
      _Float16* p = (_Float16*)C + off;
      if (beta) v += (float)*p;
      *p = (_Float16)v;
    }
  }
}

// ---------------------------------------------------------------------------
// Softmax over the 191 criss-cross logits per (n,y,x); f32 math on f16 logits.
// e_row/e_col: (N,96,H,W).  Outputs:
//   wgt_row (n,i,y,x)  and  wgt_col (n,j,x,y)  [diag j==y forced to 0]
// One block per (n,y), 96 threads = x; everything is L2-resident.
// ---------------------------------------------------------------------------
__global__ void cc_softmax(const _Float16* __restrict__ e_row,
                           const _Float16* __restrict__ e_col,
                           _Float16* __restrict__ wgt_row,
                           _Float16* __restrict__ wgt_col)
{
  const long HW = 96 * 96;
  int n = blockIdx.x / 96;
  int y = blockIdx.x % 96;
  int x = threadIdx.x;
  long br = (long)n * 96 * HW + (long)y * 96 + x;   // (n,*,y,x)

  float mx = -3.0e38f;
  for (int i = 0; i < 96; ++i) mx = fmaxf(mx, (float)e_row[br + (long)i * HW]);
  for (int j = 0; j < 96; ++j)
    if (j != y) mx = fmaxf(mx, (float)e_col[br + (long)j * HW]);

  float s = 0.f;
  for (int i = 0; i < 96; ++i) s += __expf((float)e_row[br + (long)i * HW] - mx);
  for (int j = 0; j < 96; ++j)
    if (j != y) s += __expf((float)e_col[br + (long)j * HW] - mx);
  float inv = 1.f / s;

  long wr = (long)n * 96 * HW + (long)y * 96 + x;   // (n,i,y,x)
  for (int i = 0; i < 96; ++i)
    wgt_row[wr + (long)i * HW] =
        (_Float16)(__expf((float)e_row[br + (long)i * HW] - mx) * inv);
  long wc = (long)n * 96 * HW + (long)x * 96 + y;   // (n,j,x,y)
  for (int j = 0; j < 96; ++j)
    wgt_col[wc + (long)j * HW] =
        (j == y) ? (_Float16)0.f
                 : (_Float16)(__expf((float)e_col[br + (long)j * HW] - mx) * inv);
}

// ---------------------------------------------------------------------------
template<bool BKC>
static void launch_gemm(hipStream_t stream,
                        const _Float16* A, long a_rs, long a_bso, long a_bsi,
                        const _Float16* B, long b_rs, long b_cs, long b_bso, long b_bsi,
                        void* C, long c_rs, long c_cs, long c_bso, long c_bsi,
                        int M, int Ncols, int K, int nbatch, int inner,
                        const float* bias, const float* residual,
                        int f32out, int beta)
{
  dim3 grid((unsigned)(Ncols / 32), (unsigned)(M / 32), (unsigned)nbatch);
  wmma_gemm_f16<BKC><<<grid, dim3(128), 0, stream>>>(
      A, a_rs, a_bso, a_bsi, B, b_rs, b_cs, b_bso, b_bsi,
      C, c_rs, c_cs, c_bso, c_bsi, K, inner, bias, residual, f32out, beta);
}

extern "C" void kernel_launch(void* const* d_in, const int* in_sizes, int n_in,
                              void* d_out, int out_size, void* d_ws, size_t ws_size,
                              hipStream_t stream) {
  (void)in_sizes; (void)n_in; (void)out_size; (void)ws_size;
  const long N = 8, C_IN = 512, C_INR = 64, C_OUT = 512, H = 96, W = 96;
  const long HW = H * W;   // 9216

  const float* x     = (const float*)d_in[0];
  const float* t_w   = (const float*)d_in[1];
  const float* t_b   = (const float*)d_in[2];
  const float* f_w   = (const float*)d_in[3];
  const float* f_b   = (const float*)d_in[4];
  const float* g_w   = (const float*)d_in[5];
  const float* g_b   = (const float*)d_in[6];
  const float* inc_w = (const float*)d_in[7];
  const float* inc_b = (const float*)d_in[8];
  float* out = (float*)d_out;

  // Workspace (~300 MB); outmap aliases x_h (dead after the conv GEMMs).
  char* ws = (char*)d_ws;
  size_t off = 0;
  auto carve = [&](size_t bytes) {
    void* p = ws + off; off += (bytes + 255) & ~(size_t)255; return p;
  };
  _Float16* x_h     = (_Float16*)carve((size_t)(N * C_IN * HW) * 2);
  _Float16* tw_h    = (_Float16*)carve((size_t)(C_INR * C_IN) * 2);
  _Float16* fw_h    = (_Float16*)carve((size_t)(C_INR * C_IN) * 2);
  _Float16* gw_h    = (_Float16*)carve((size_t)(C_OUT * C_IN) * 2);
  _Float16* incw_h  = (_Float16*)carve((size_t)(C_IN * C_OUT) * 2);
  _Float16* t_h     = (_Float16*)carve((size_t)(N * HW * C_INR) * 2); // (n,y,x,c)
  _Float16* f_h     = (_Float16*)carve((size_t)(N * HW * C_INR) * 2); // (n,y,x,c)
  _Float16* g_row   = (_Float16*)carve((size_t)(N * C_OUT * HW) * 2); // (n,c,y,x)
  _Float16* g_col   = (_Float16*)carve((size_t)(N * C_OUT * HW) * 2); // (n,c,x,y)
  _Float16* e_row   = (_Float16*)carve((size_t)(N * 96 * HW) * 2);    // (n,i,y,x)
  _Float16* e_col   = (_Float16*)carve((size_t)(N * 96 * HW) * 2);    // (n,j,y,x)
  _Float16* wgt_row = (_Float16*)carve((size_t)(N * 96 * HW) * 2);    // (n,i,y,x)
  _Float16* wgt_col = (_Float16*)carve((size_t)(N * 96 * HW) * 2);    // (n,j,x,y)
  _Float16* outmap  = x_h;                                            // (n,c,y,x)

  auto cvt = [&](const float* src, _Float16* dst, long n) {
    cast_f32_to_f16<<<dim3((unsigned)((n + 255) / 256)), dim3(256), 0, stream>>>(src, dst, n);
  };

  // 1) casts
  cvt(x,     x_h,    N * C_IN * HW);
  cvt(t_w,   tw_h,   C_INR * C_IN);
  cvt(f_w,   fw_h,   C_INR * C_IN);
  cvt(g_w,   gw_h,   C_OUT * C_IN);
  cvt(inc_w, incw_h, C_IN * C_OUT);

  // 2) t/f convs, channel-last output t_h/f_h (n,p,c):  C[m=c][p] strided.
  launch_gemm<false>(stream, tw_h, C_IN, 0, 0,
                     x_h, HW, 1, C_IN * HW, 0,
                     t_h, 1, C_INR, HW * C_INR, 0,
                     (int)C_INR, (int)HW, (int)C_IN, (int)N, 1, t_b, nullptr, 0, 0);
  launch_gemm<false>(stream, fw_h, C_IN, 0, 0,
                     x_h, HW, 1, C_IN * HW, 0,
                     f_h, 1, C_INR, HW * C_INR, 0,
                     (int)C_INR, (int)HW, (int)C_IN, (int)N, 1, f_b, nullptr, 0, 0);

  // 3) g conv twice: (n,c,y,x) and (n,c,x,y)
  launch_gemm<false>(stream, gw_h, C_IN, 0, 0,
                     x_h, HW, 1, C_IN * HW, 0,
                     g_row, HW, 1, C_OUT * HW, 0,
                     (int)C_OUT, (int)HW, (int)C_IN, (int)N, 1, g_b, nullptr, 0, 0);
  launch_gemm<false>(stream, gw_h, C_IN, 0, 0,           // batch (n,y), cols = x
                     x_h, HW, 1, C_IN * HW, W,
                     g_col, HW, H, C_OUT * HW, 1,
                     (int)C_OUT, (int)W, (int)C_IN, (int)(N * H), (int)H,
                     g_b, nullptr, 0, 0);

  // 4) row energies, batch (n,y): E[x,i] = sum_c t[y,x,c] * f[y,i,c]
  launch_gemm<true>(stream, t_h, C_INR, HW * C_INR, W * C_INR,
                    f_h, 1, C_INR, HW * C_INR, W * C_INR,
                    e_row, 1, HW, 96 * HW, W,
                    96, 96, (int)C_INR, (int)(N * H), (int)H, nullptr, nullptr, 0, 0);

  // 5) col energies, batch (n,x): E[y,j] = sum_c t[y,x,c] * f[j,x,c]
  launch_gemm<true>(stream, t_h, W * C_INR, HW * C_INR, C_INR,
                    f_h, 1, W * C_INR, HW * C_INR, C_INR,
                    e_col, W, HW, 96 * HW, 1,
                    96, 96, (int)C_INR, (int)(N * W), (int)W, nullptr, nullptr, 0, 0);

  // 6) softmax -> wgt_row (n,i,y,x), wgt_col (n,j,x,y)
  cc_softmax<<<dim3((unsigned)(N * H)), dim3(96), 0, stream>>>(e_row, e_col,
                                                               wgt_row, wgt_col);

  // 7) row aggregation, batch (n,y): out[c,x] = sum_i g_row[c,y,i] * wgt_row[i,y,x]
  launch_gemm<false>(stream, g_row, HW, C_OUT * HW, W,
                     wgt_row, HW, 1, 96 * HW, W,
                     outmap, HW, 1, C_OUT * HW, W,
                     (int)C_OUT, (int)W, 96, (int)(N * H), (int)H, nullptr, nullptr, 0, 0);

  // 8) col aggregation (accumulate), batch (n,x):
  //    out[c,y] += sum_j g_col[c,x,j] * wgt_col[j,x,y]
  launch_gemm<false>(stream, g_col, HW, C_OUT * HW, H,
                     wgt_col, HW, 1, 96 * HW, H,
                     outmap, HW, W, C_OUT * HW, 1,
                     (int)C_OUT, (int)H, 96, (int)(N * W), (int)W, nullptr, nullptr, 0, 1);

  // 9) final 1x1 conv + bias + residual: out = x + inc_b + inc_w * outmap
  launch_gemm<false>(stream, incw_h, C_OUT, 0, 0,
                     outmap, HW, 1, C_OUT * HW, 0,
                     out, HW, 1, C_IN * HW, 0,
                     (int)C_IN, (int)HW, (int)C_OUT, (int)N, 1, inc_b, x, 1, 0);
}
